// MistralAttention_73538430043011
// MI455X (gfx1250) — matrix operand order repaired and checked by reading, not compiled, not run on hardware
//
#include <hip/hip_runtime.h>
#include <hip/hip_bf16.h>

// ---------------------------------------------------------------------------
// MI455X (gfx1250) Mistral attention block.
// v2: hoist all f32->bf16 conversion out of the GEMM inner loops (one-shot
// bandwidth-bound passes), and widen the GEMM tile to 32x64 per wave so the
// inner loop is pure global_load_b128 + v_wmma_f32_16x16x32_bf16.
// Workload is compute-bound (~240 GFLOP vs ~10us of HBM traffic at 23.3TB/s),
// so WMMA-pipe density is the objective. wave32 layouts per
// CDNA5 ISA section 7.12.2.
// ---------------------------------------------------------------------------

typedef __attribute__((ext_vector_type(16))) __bf16 v16bf;
typedef __attribute__((ext_vector_type(8)))  __bf16 v8bf;
typedef __attribute__((ext_vector_type(8)))  float  v8f;

#define S_LEN   2048
#define HID     4096
#define NH      32
#define NKV     8
#define DHEAD   128
#define WINDOW  1024

__device__ __forceinline__ v16bf cat8(v8bf lo, v8bf hi) {
  return __builtin_shufflevector(lo, hi, 0,1,2,3,4,5,6,7,8,9,10,11,12,13,14,15);
}

__device__ __forceinline__ v16bf frag_from_bf16(const __bf16* p, int off0, int off1) {
  return cat8(*(const v8bf*)(p + off0), *(const v8bf*)(p + off1));
}

__device__ __forceinline__ v8f wmma_bf16(v16bf a, v16bf b, v8f c) {
  return __builtin_amdgcn_wmma_f32_16x16x32_bf16(
      /*neg_a=*/false, a, /*neg_b=*/false, b,
      /*c_mod=*/(short)0, c, /*reuse_a=*/false, /*reuse_b=*/false);
}

// ---------------------------------------------------------------------------
// One-shot f32 -> bf16 downconvert (8 elements / thread, vector ld/st).
// ---------------------------------------------------------------------------
__global__ void cvt_bf16_kernel(const float* __restrict__ in,
                                __bf16* __restrict__ outp, int n8) {
  int idx = blockIdx.x * blockDim.x + threadIdx.x;
  if (idx >= n8) return;
  v8f f = *(const v8f*)(in + (size_t)idx * 8);
  v8bf o;
#pragma unroll
  for (int i = 0; i < 8; ++i) o[i] = (__bf16)f[i];
  *(v8bf*)(outp + (size_t)idx * 8) = o;
}

// ---------------------------------------------------------------------------
// GEMM: C[M,N] (f32) = A[M,K] * W[N,K]^T, all-bf16 operands, f32 accumulate.
// One wave per 32x64 tile: 2 A-frags x 4 B-frags -> 8 WMMAs per 32-deep
// k-step, pure b128 loads in the loop (no conversion VALU).
// A layout: lane m / m+16 holds row m, K chunks {k+8h, k+16+8h}.
// B layout: lane n / n+16 holds W row n (= B column n), K chunks {k+8h, k+16+8h}.
// ---------------------------------------------------------------------------
__global__ __launch_bounds__(32)
void gemm_bf16_kernel(const __bf16* __restrict__ A, const __bf16* __restrict__ W,
                      float* __restrict__ C, int N, int K) {
  const int lane = threadIdx.x & 31;
  const int half = lane >> 4;
  const int lr   = lane & 15;
  const int n0   = blockIdx.x * 64;
  const int m0   = blockIdx.y * 32;

  v8f acc[2][4] = {};
  for (int k = 0; k < K; k += 32) {
    v16bf a0 = frag_from_bf16(A + (size_t)(m0 + lr) * K + k,      8 * half, 16 + 8 * half);
    v16bf a1 = frag_from_bf16(A + (size_t)(m0 + 16 + lr) * K + k, 8 * half, 16 + 8 * half);
#pragma unroll
    for (int t = 0; t < 4; ++t) {
      v16bf b = frag_from_bf16(W + (size_t)(n0 + t * 16 + lr) * K + k,
                               8 * half, 16 + 8 * half);
      acc[0][t] = wmma_bf16(a0, b, acc[0][t]);
      acc[1][t] = wmma_bf16(a1, b, acc[1][t]);
    }
  }
  // C/D layout: VGPR r, lanes 0-15 -> (M=r, N=lane); lanes 16-31 -> (M=r+8).
#pragma unroll
  for (int mt = 0; mt < 2; ++mt)
#pragma unroll
    for (int t = 0; t < 4; ++t)
#pragma unroll
      for (int r = 0; r < 8; ++r)
        C[(size_t)(m0 + 16 * mt + r + 8 * half) * N + (n0 + t * 16 + lr)] =
            acc[mt][t][r];
}

// ---------------------------------------------------------------------------
// RoPE: f32 in -> rotated bf16 out. nh = 32 (Q) or 8 (K).
// ---------------------------------------------------------------------------
__global__ void rope_bf16_kernel(const float* __restrict__ in,
                                 const int* __restrict__ pos,
                                 __bf16* __restrict__ outp, int nh) {
  int idx = blockIdx.x * blockDim.x + threadIdx.x;
  int j = idx & 63;
  int h = (idx >> 6) % nh;
  int s = idx / (nh << 6);
  if (s >= S_LEN) return;
  const int W = nh * DHEAD;
  float p   = (float)pos[s];
  // inv_freq = 10000^(-2j/128) = exp(-ln(1e4) * 2j/128)
  float inv = __expf(-9.210340371976184f * (float)(2 * j) * (1.0f / 128.0f));
  float ang = p * inv;
  float c = __cosf(ang), sn = __sinf(ang);
  size_t base = (size_t)s * W + h * DHEAD;
  float x1 = in[base + j];
  float x2 = in[base + 64 + j];
  outp[base + j]      = (__bf16)(x1 * c - x2 * sn);
  outp[base + 64 + j] = (__bf16)(x2 * c + x1 * sn);
}

// V: [s][kvh*128+d] f32 -> Vt[kvh*128+d][s] bf16 (so PV B-frags are contiguous)
__global__ void vtrans_kernel(const float* __restrict__ Vf, __bf16* __restrict__ Vt) {
  int idx = blockIdx.x * blockDim.x + threadIdx.x;
  if (idx >= S_LEN * NKV * DHEAD) return;
  int s = idx >> 10;        // / (NKV*DHEAD)=1024
  int c = idx & 1023;
  Vt[(size_t)c * S_LEN + s] = (__bf16)Vf[idx];
}

// ---------------------------------------------------------------------------
// Windowed-causal GQA flash attention. One wave per (head, 16-query tile).
// Online softmax; 32 keys per step: 8 QK^T WMMAs + 8 P*V WMMAs.
// ---------------------------------------------------------------------------
__global__ __launch_bounds__(32)
void attn_kernel(const __bf16* __restrict__ Q, const __bf16* __restrict__ Kb,
                 const __bf16* __restrict__ Vt, __bf16* __restrict__ Ctx) {
  const int h    = blockIdx.y;
  const int q0   = blockIdx.x * 16;
  const int kvh  = h >> 2;                     // GROUPS = 4
  const int lane = threadIdx.x & 31;
  const int half = lane >> 4;
  const int lr   = lane & 15;
  const float scale = 0.08838834764831845f;    // 1/sqrt(128)

  __shared__ __bf16 sP[16 * 32];               // P tile bounce: C-layout -> A-layout

  // Preload 4 Q A-fragments covering d = 0..127 for rows q0..q0+15.
  v16bf qf[4];
  const __bf16* qrow = Q + (size_t)(q0 + lr) * HID + h * DHEAD;
#pragma unroll
  for (int j = 0; j < 4; ++j)
    qf[j] = frag_from_bf16(qrow + 32 * j, 16 * half, 16 * half + 8);

  float m_row[8], l_row[8];
  v8f acc[8] = {};
#pragma unroll
  for (int r = 0; r < 8; ++r) { m_row[r] = -1e30f; l_row[r] = 0.0f; }

  const int kstart = (q0 - (WINDOW - 16) > 0) ? (q0 - (WINDOW - 16)) : 0;
  const int kend   = q0 + 16;

  for (int k0 = kstart; k0 < kend; k0 += 32) {
    // ---- scores for two 16-key tiles -------------------------------------
    v8f s0 = {}, s1 = {};
#pragma unroll
    for (int j = 0; j < 4; ++j) {
      const __bf16* pk0 = Kb + (size_t)(k0 + lr) * (NKV * DHEAD) + kvh * DHEAD + 32 * j;
      const __bf16* pk1 = Kb + (size_t)(k0 + 16 + lr) * (NKV * DHEAD) + kvh * DHEAD + 32 * j;
      s0 = wmma_bf16(qf[j], frag_from_bf16(pk0, 16 * half, 16 * half + 8), s0);
      s1 = wmma_bf16(qf[j], frag_from_bf16(pk1, 16 * half, 16 * half + 8), s1);
    }
    // ---- masked online softmax (rows span the 16 lanes of each half) -----
#pragma unroll
    for (int r = 0; r < 8; ++r) {
      int row = q0 + r + 8 * half;
      int c0 = k0 + lr, c1 = k0 + 16 + lr;
      float v0 = s0[r] * scale;
      float v1 = s1[r] * scale;
      if (c0 > row || row - c0 >= WINDOW) v0 = -1e30f;
      if (c1 > row || row - c1 >= WINDOW) v1 = -1e30f;
      float tmax = fmaxf(v0, v1);
#pragma unroll
      for (int off = 1; off < 16; off <<= 1)
        tmax = fmaxf(tmax, __shfl_xor(tmax, off, 16));
      float mn = fmaxf(m_row[r], tmax);
      float e0 = __expf(v0 - mn);
      float e1 = __expf(v1 - mn);
      float corr = __expf(m_row[r] - mn);
      float sum = e0 + e1;
#pragma unroll
      for (int off = 1; off < 16; off <<= 1)
        sum += __shfl_xor(sum, off, 16);
      l_row[r] = l_row[r] * corr + sum;
      m_row[r] = mn;
#pragma unroll
      for (int t = 0; t < 8; ++t) acc[t][r] *= corr;
      sP[(r + 8 * half) * 32 + lr]      = (__bf16)e0;
      sP[(r + 8 * half) * 32 + 16 + lr] = (__bf16)e1;
    }
    __syncthreads();
    // Re-read P in A-fragment layout (lane m: row m, K chunks {8h, 16+8h}).
    v16bf pa = frag_from_bf16(&sP[lr * 32], 8 * half, 16 + 8 * half);
    __syncthreads();
    // ---- P * V -----------------------------------------------------------
#pragma unroll
    for (int t = 0; t < 8; ++t) {
      const __bf16* pv = Vt + (size_t)(kvh * DHEAD + t * 16 + lr) * S_LEN + k0;
      v16bf vb = frag_from_bf16(pv, 8 * half, 16 + 8 * half);
      acc[t] = wmma_bf16(pa, vb, acc[t]);
    }
  }

  // ---- epilogue: normalize, store context bf16 [s][h*128+d] --------------
#pragma unroll
  for (int t = 0; t < 8; ++t)
#pragma unroll
    for (int r = 0; r < 8; ++r) {
      float o = acc[t][r] / l_row[r];
      Ctx[(size_t)(q0 + r + 8 * half) * HID + h * DHEAD + t * 16 + lr] = (__bf16)o;
    }
}

// ---------------------------------------------------------------------------
// Host-side launch. Inputs: hidden_states, attention_mask(unused; mask is
// analytic), position_ids, Wq, Wk, Wv, Wo. Output: f32 [2048,4096].
// ---------------------------------------------------------------------------
extern "C" void kernel_launch(void* const* d_in, const int* in_sizes, int n_in,
                              void* d_out, int out_size, void* d_ws, size_t ws_size,
                              hipStream_t stream) {
  (void)in_sizes; (void)n_in; (void)out_size; (void)ws_size;
  const float* X   = (const float*)d_in[0];
  const int*   pos = (const int*)d_in[2];
  const float* Wq  = (const float*)d_in[3];
  const float* Wk  = (const float*)d_in[4];
  const float* Wv  = (const float*)d_in[5];
  const float* Wo  = (const float*)d_in[6];
  float* out = (float*)d_out;

  char* w = (char*)d_ws;
  float*  Qf   = (float*)(w + 0);                 // 32 MB (reused as Ctx later)
  float*  Kf   = (float*)(w + (33ull << 20));     //  8 MB
  float*  Vf   = (float*)(w + (41ull << 20));     //  8 MB
  __bf16* Qbf  = (__bf16*)(w + (49ull << 20));    // 16 MB
  __bf16* Kbf  = (__bf16*)(w + (65ull << 20));    //  4 MB
  __bf16* Vtb  = (__bf16*)(w + (69ull << 20));    //  4 MB
  __bf16* Xbf  = (__bf16*)(w + (73ull << 20));    // 16 MB
  __bf16* Wqb  = (__bf16*)(w + (89ull << 20));    // 32 MB
  __bf16* Wkb  = (__bf16*)(w + (121ull << 20));   //  8 MB
  __bf16* Wvb  = (__bf16*)(w + (129ull << 20));   //  8 MB
  __bf16* Wob  = (__bf16*)(w + (137ull << 20));   // 32 MB
  __bf16* Ctx  = (__bf16*)(w + 0);                // overlays dead Qf

  dim3 wave(32);
  // --- one-shot f32 -> bf16 downconverts (bandwidth-bound, ~11us total) ---
  const int XN  = S_LEN * HID / 8;
  const int WQN = HID * HID / 8;
  const int WKN = NKV * DHEAD * HID / 8;
  cvt_bf16_kernel<<<(XN  + 255) / 256, 256, 0, stream>>>(X,  Xbf, XN);
  cvt_bf16_kernel<<<(WQN + 255) / 256, 256, 0, stream>>>(Wq, Wqb, WQN);
  cvt_bf16_kernel<<<(WKN + 255) / 256, 256, 0, stream>>>(Wk, Wkb, WKN);
  cvt_bf16_kernel<<<(WKN + 255) / 256, 256, 0, stream>>>(Wv, Wvb, WKN);
  cvt_bf16_kernel<<<(WQN + 255) / 256, 256, 0, stream>>>(Wo, Wob, WQN);
  // --- Q/K/V projections (pure bf16 WMMA inner loops) ---------------------
  gemm_bf16_kernel<<<dim3(HID / 64, S_LEN / 32), wave, 0, stream>>>(Xbf, Wqb, Qf, HID, HID);
  gemm_bf16_kernel<<<dim3((NKV * DHEAD) / 64, S_LEN / 32), wave, 0, stream>>>(Xbf, Wkb, Kf, NKV * DHEAD, HID);
  gemm_bf16_kernel<<<dim3((NKV * DHEAD) / 64, S_LEN / 32), wave, 0, stream>>>(Xbf, Wvb, Vf, NKV * DHEAD, HID);
  // --- RoPE + downconvert, V transpose ------------------------------------
  rope_bf16_kernel<<<(S_LEN * NH * 64) / 256, 256, 0, stream>>>(Qf, pos, Qbf, NH);
  rope_bf16_kernel<<<(S_LEN * NKV * 64) / 256, 256, 0, stream>>>(Kf, pos, Kbf, NKV);
  vtrans_kernel<<<(S_LEN * NKV * DHEAD) / 256, 256, 0, stream>>>(Vf, Vtb);
  // --- attention (windowed causal, GQA) -----------------------------------
  attn_kernel<<<dim3(S_LEN / 16, NH), wave, 0, stream>>>(Qbf, Kbf, Vtb, Ctx);
  // --- output projection ---------------------------------------------------
  gemm_bf16_kernel<<<dim3(HID / 64, S_LEN / 32), wave, 0, stream>>>(Ctx, Wob, out, HID, HID);
}
